// GraphAttentionLayer_15779709846319
// MI455X (gfx1250) — compile-verified, matrix operand order checked by reading, not run on hardware
//
#include <hip/hip_runtime.h>
#include <hip/hip_bf16.h>
#include <cstdint>

typedef __attribute__((ext_vector_type(16))) __bf16 v16bf;
typedef __attribute__((ext_vector_type(8)))  float  v8f;
typedef __attribute__((ext_vector_type(4)))  int    vi4;

union BFrag16 { v16bf v; uint4 q[2]; };          // one WMMA A/B fragment (16 bf16 = 32B)
union BPack16 { __bf16 h[16]; uint4 q[2]; };
union BPack32 { __bf16 h[32]; uint4 q[4]; };
union BPack8  { __bf16 h[8];  uint4 q; };

__device__ __forceinline__ float leaky(float x) { return x >= 0.f ? x : 0.2f * x; }

__device__ __forceinline__ v8f vzero()
{
    v8f z;
    #pragma unroll
    for (int i = 0; i < 8; ++i) z[i] = 0.f;
    return z;
}

// ---------------------------------------------------------------------------
// Kernel 1: Wh = h @ W^T, stored transposed as bf16: whT[b][o][n]
// block = 256 thr (8 waves), each wave does a 16-row x 256-col strip.
// ---------------------------------------------------------------------------
__global__ __launch_bounds__(256)
void wh_gemm_kernel(const float* __restrict__ h, const float* __restrict__ W,
                    __bf16* __restrict__ whT)
{
    __shared__ __bf16 Wlds[256][32];                 // W[:, kb..kb+31] as bf16, 16 KB

    const int tid   = threadIdx.x;
    const int lane  = tid & 31;
    const int wv    = tid >> 5;
    const int lh    = lane & 15;
    const int hh    = lane >> 4;
    const int koff  = hh << 3;                       // 0 or 8
    const int rbase = blockIdx.x * 128 + wv * 16;    // flattened row (b*2048+n)
    const int b     = rbase >> 11;
    const int nbase = rbase & 2047;

    v8f acc[16];
    #pragma unroll
    for (int t = 0; t < 16; ++t) acc[t] = vzero();

    const float* hrow = h + (size_t)(rbase + lh) * 256;

    for (int kb = 0; kb < 256; kb += 32) {
        // cooperative stage: thread o converts W[o][kb..kb+31] -> bf16
        {
            const float4* wp = reinterpret_cast<const float4*>(W + tid * 256 + kb);
            BPack32 pk;
            #pragma unroll
            for (int q = 0; q < 8; ++q) {
                float4 f = wp[q];
                pk.h[4*q+0] = (__bf16)f.x; pk.h[4*q+1] = (__bf16)f.y;
                pk.h[4*q+2] = (__bf16)f.z; pk.h[4*q+3] = (__bf16)f.w;
            }
            uint4* dst = reinterpret_cast<uint4*>(&Wlds[tid][0]);
            #pragma unroll
            for (int q = 0; q < 4; ++q) dst[q] = pk.q[q];
        }
        __syncthreads();

        // A fragment: 16x32 of h, rows via lanes, K halves via lane-half
        BFrag16 fa;
        {
            const float4* a0 = reinterpret_cast<const float4*>(hrow + kb + koff);
            const float4* a1 = reinterpret_cast<const float4*>(hrow + kb + koff + 16);
            float4 f0 = a0[0], f1 = a0[1], f2 = a1[0], f3 = a1[1];
            BPack16 pk;
            pk.h[0]=(__bf16)f0.x; pk.h[1]=(__bf16)f0.y; pk.h[2]=(__bf16)f0.z; pk.h[3]=(__bf16)f0.w;
            pk.h[4]=(__bf16)f1.x; pk.h[5]=(__bf16)f1.y; pk.h[6]=(__bf16)f1.z; pk.h[7]=(__bf16)f1.w;
            pk.h[8]=(__bf16)f2.x; pk.h[9]=(__bf16)f2.y; pk.h[10]=(__bf16)f2.z; pk.h[11]=(__bf16)f2.w;
            pk.h[12]=(__bf16)f3.x; pk.h[13]=(__bf16)f3.y; pk.h[14]=(__bf16)f3.z; pk.h[15]=(__bf16)f3.w;
            fa.q[0] = pk.q[0]; fa.q[1] = pk.q[1];
        }

        #pragma unroll
        for (int t = 0; t < 16; ++t) {
            const uint4* bp = reinterpret_cast<const uint4*>(&Wlds[t * 16 + lh][koff]);
            BFrag16 fb;
            fb.q[0] = bp[0];      // K = koff+0..7
            fb.q[1] = bp[2];      // K = koff+16..23
            acc[t] = __builtin_amdgcn_wmma_f32_16x16x32_bf16(
                         false, fa.v, false, fb.v, (short)0, acc[t], false, false);
        }
        __syncthreads();
    }

    // store transposed: whT[b][col][n], 8 consecutive n per lane -> b128 store
    #pragma unroll
    for (int t = 0; t < 16; ++t) {
        BPack8 pk;
        #pragma unroll
        for (int k = 0; k < 8; ++k) pk.h[k] = (__bf16)acc[t][k];
        const int col = t * 16 + lh;
        uint4* op = reinterpret_cast<uint4*>(
            whT + (((size_t)(b * 256 + col)) << 11) + nbase + hh * 8);
        *op = pk.q;
    }
}

// ---------------------------------------------------------------------------
// Kernel 2: s1/s2 = Wh @ a1/a2, plus per-batch max(s2) (deterministic uint max)
// ---------------------------------------------------------------------------
__global__ __launch_bounds__(256)
void s_kernel(const __bf16* __restrict__ whT, const float* __restrict__ a,
              float* __restrict__ s1, float* __restrict__ s2,
              unsigned* __restrict__ s2max)
{
    __shared__ float red[256];
    const int tid = threadIdx.x;
    const int ng  = blockIdx.x * 256 + tid;          // flattened node
    const int b   = ng >> 11;
    const int n   = ng & 2047;

    const __bf16* base = whT + (((size_t)b) << 19) + n;   // b*256*2048
    float a1v = 0.f, a2v = 0.f;
    for (int o = 0; o < 256; ++o) {
        float w = (float)base[((size_t)o) << 11];
        a1v += w * a[o];
        a2v += w * a[256 + o];
    }
    s1[ng] = a1v;
    s2[ng] = a2v;

    red[tid] = a2v;
    __syncthreads();
    for (int s = 128; s > 0; s >>= 1) {
        if (tid < s) red[tid] = fmaxf(red[tid], red[tid + s]);
        __syncthreads();
    }
    if (tid == 0) {
        unsigned u = __float_as_uint(red[0]);
        u = (u & 0x80000000u) ? ~u : (u | 0x80000000u);   // order-preserving encode
        atomicMax(&s2max[b], u);
    }
}

// ---------------------------------------------------------------------------
// Kernel 3: fused masked-softmax + (P @ Wh) + ELU.
// block = 256 thr (8 waves) -> 32 rows x 256 cols; j loop in 128-wide tiles.
// Shift M_i = lrelu(s1_i + max_j s2_j) >= true row max  => single pass, no rescale.
// adj is a one-shot 134 MB stream -> non-temporal loads keep L2 for WhT reuse.
// ---------------------------------------------------------------------------
__global__ __launch_bounds__(256)
void gat_attn_kernel(const int* __restrict__ adj, const __bf16* __restrict__ whT,
                     const float* __restrict__ s1, const float* __restrict__ s2,
                     const unsigned* __restrict__ s2max, float* __restrict__ out)
{
    __shared__ __bf16 P[32][136];                    // P tile, padded row stride
    __shared__ float  lsum[32][8];
    __shared__ float  l_lds[32];
    __shared__ float  s1_lds[32];
    __shared__ float  M_lds[32];

    const int tid  = threadIdx.x;
    const int lane = tid & 31;
    const int wv   = tid >> 5;
    const int lh   = lane & 15;
    const int hh   = lane >> 4;
    const int koff = hh << 3;
    const int b    = blockIdx.x >> 6;
    const int i0   = (blockIdx.x & 63) << 5;

    if (tid < 32) {
        unsigned u = s2max[b];
        float m2 = __uint_as_float((u & 0x80000000u) ? (u & 0x7FFFFFFFu) : ~u);
        float sv = s1[b * 2048 + i0 + tid];
        s1_lds[tid] = sv;
        M_lds[tid]  = leaky(sv + m2);
        l_lds[tid]  = 0.f;
    }
    __syncthreads();

    v8f cc[2][2];
    #pragma unroll
    for (int mt = 0; mt < 2; ++mt)
        #pragma unroll
        for (int nt = 0; nt < 2; ++nt) cc[mt][nt] = vzero();

    const int il  = tid >> 3;                        // row 0..31
    const int seg = tid & 7;                         // 16-wide j segment
    const size_t adjrow = ((size_t)(b * 2048 + i0 + il)) * 2048 + seg * 16;
    const float* s2row  = s2 + b * 2048 + seg * 16;
    const int c0 = wv * 32 + lh;

    for (int j0 = 0; j0 < 2048; j0 += 128) {
        // --- phase A: P tile = masked exp(e - M), partial denominators ---
        {
            const vi4*    ap = reinterpret_cast<const vi4*>(adj + adjrow + j0);
            const float4* sp = reinterpret_cast<const float4*>(s2row + j0);
            const float sv = s1_lds[il], Mv = M_lds[il];
            float partial = 0.f;
            BPack16 pk;
            #pragma unroll
            for (int q = 0; q < 4; ++q) {
                vi4 av = __builtin_nontemporal_load(ap + q);   // streaming: TH=NT
                float4 f = sp[q];
                float e, p;
                e = leaky(sv + f.x); p = av[0] ? __expf(e - Mv) : 0.f; partial += p; pk.h[4*q+0] = (__bf16)p;
                e = leaky(sv + f.y); p = av[1] ? __expf(e - Mv) : 0.f; partial += p; pk.h[4*q+1] = (__bf16)p;
                e = leaky(sv + f.z); p = av[2] ? __expf(e - Mv) : 0.f; partial += p; pk.h[4*q+2] = (__bf16)p;
                e = leaky(sv + f.w); p = av[3] ? __expf(e - Mv) : 0.f; partial += p; pk.h[4*q+3] = (__bf16)p;
            }
            lsum[il][seg] = partial;
            uint4* pd = reinterpret_cast<uint4*>(&P[il][seg * 16]);
            pd[0] = pk.q[0]; pd[1] = pk.q[1];
        }
        __syncthreads();

        // deterministic denominator accumulation (wave 0, fixed order)
        if (tid < 32) {
            float s = 0.f;
            #pragma unroll
            for (int q = 0; q < 8; ++q) s += lsum[tid][q];
            l_lds[tid] += s;
        }

        // --- phase B: cc += P(32x128) @ WhT-slice(128x32 per wave) ---
        #pragma unroll
        for (int kk = 0; kk < 4; ++kk) {
            BFrag16 fa0, fa1, fb0, fb1;
            const uint4* p0 = reinterpret_cast<const uint4*>(&P[lh][kk * 32 + koff]);
            fa0.q[0] = p0[0]; fa0.q[1] = p0[2];
            const uint4* p1 = reinterpret_cast<const uint4*>(&P[16 + lh][kk * 32 + koff]);
            fa1.q[0] = p1[0]; fa1.q[1] = p1[2];

            const size_t jb = (size_t)(j0 + kk * 32 + koff);
            const uint4* g0 = reinterpret_cast<const uint4*>(
                whT + (((size_t)(b * 256 + c0)) << 11) + jb);
            fb0.q[0] = g0[0]; fb0.q[1] = g0[2];
            const uint4* g1 = reinterpret_cast<const uint4*>(
                whT + (((size_t)(b * 256 + c0 + 16)) << 11) + jb);
            fb1.q[0] = g1[0]; fb1.q[1] = g1[2];

            cc[0][0] = __builtin_amdgcn_wmma_f32_16x16x32_bf16(false, fa0.v, false, fb0.v, (short)0, cc[0][0], false, false);
            cc[0][1] = __builtin_amdgcn_wmma_f32_16x16x32_bf16(false, fa0.v, false, fb1.v, (short)0, cc[0][1], false, false);
            cc[1][0] = __builtin_amdgcn_wmma_f32_16x16x32_bf16(false, fa1.v, false, fb0.v, (short)0, cc[1][0], false, false);
            cc[1][1] = __builtin_amdgcn_wmma_f32_16x16x32_bf16(false, fa1.v, false, fb1.v, (short)0, cc[1][1], false, false);
        }
        __syncthreads();
    }

    // --- epilogue: divide by denominator, ELU, non-temporal fp32 stores ---
    #pragma unroll
    for (int mt = 0; mt < 2; ++mt) {
        #pragma unroll
        for (int nt = 0; nt < 2; ++nt) {
            const int col = wv * 32 + nt * 16 + lh;
            #pragma unroll
            for (int k = 0; k < 8; ++k) {
                const int row = mt * 16 + koff + k;
                float v = cc[mt][nt][k] / l_lds[row];
                v = v > 0.f ? v : (__expf(v) - 1.f);
                __builtin_nontemporal_store(
                    v, out + ((size_t)(b * 2048 + i0 + row)) * 256 + col);
            }
        }
    }
}

// ---------------------------------------------------------------------------
extern "C" void kernel_launch(void* const* d_in, const int* in_sizes, int n_in,
                              void* d_out, int out_size, void* d_ws, size_t ws_size,
                              hipStream_t stream)
{
    (void)in_sizes; (void)n_in; (void)out_size; (void)ws_size;
    const float* h   = (const float*)d_in[0];
    const int*   adj = (const int*)d_in[1];
    const float* W   = (const float*)d_in[2];
    const float* a   = (const float*)d_in[3];
    float* out = (float*)d_out;

    char* ws = (char*)d_ws;
    __bf16*   whT = (__bf16*)ws;                               // 8 MB: [8][256][2048] bf16
    float*    s1  = (float*)(ws + 8388608);                    // 64 KB
    float*    s2  = (float*)(ws + 8388608 + 65536);            // 64 KB
    unsigned* s2m = (unsigned*)(ws + 8388608 + 131072);        // 32 B

    (void)hipMemsetAsync(s2m, 0, 8 * sizeof(unsigned), stream);
    hipLaunchKernelGGL(wh_gemm_kernel,  dim3(128), dim3(256), 0, stream, h, W, whT);
    hipLaunchKernelGGL(s_kernel,        dim3(64),  dim3(256), 0, stream, whT, a, s1, s2, s2m);
    hipLaunchKernelGGL(gat_attn_kernel, dim3(512), dim3(256), 0, stream, adj, whT, s1, s2, s2m, out);
}
